// StratumBlock_48318382080024
// MI455X (gfx1250) — matrix operand order repaired and checked by reading, not compiled, not run on hardware
//
#include <hip/hip_runtime.h>
#include <hip/hip_bf16.h>

// ---------------------------------------------------------------------------
// MI455X (gfx1250) implementation: bf16 WMMA GEMMs + flash attention.
// All matrix math via v_wmma_f32_16x16x32_bf16 (wave32, 16x16 tiles).
// ---------------------------------------------------------------------------

typedef __attribute__((ext_vector_type(16))) __bf16 v16bf;
typedef __attribute__((ext_vector_type(8)))  float  v8f;

#define C_CH  128
#define HWPX  9216
#define WIMG  96
#define PADW  98

union FragU  { uint4 q[2]; v16bf v; };
union Pack8U { __bf16 h[8]; uint4 q; };

__device__ inline v16bf ld_frag_contig(const __bf16* p) {
    FragU u; const uint4* q = (const uint4*)p;
    u.q[0] = q[0]; u.q[1] = q[1]; return u.v;
}
__device__ inline v16bf ld_frag_split(const __bf16* p0, const __bf16* p1) {
    FragU u; u.q[0] = *(const uint4*)p0; u.q[1] = *(const uint4*)p1; return u.v;
}
__device__ inline v8f wmma_bf16(v16bf a, v16bf b, v8f c) {
    return __builtin_amdgcn_wmma_f32_16x16x32_bf16(false, a, false, b,
                                                   (short)0, c, false, false);
}
__device__ inline float redmax16(float v) {
    #pragma unroll
    for (int off = 8; off > 0; off >>= 1) v = fmaxf(v, __shfl_xor(v, off, 16));
    return v;
}

// ---------------------------------------------------------------------------
// Prep kernels
// ---------------------------------------------------------------------------

// x (C,HW) f32  ->  xb (HW,C) bf16   (pixel-major so B/A fragments are b128s)
__global__ void k_convert_x(const float* __restrict__ x, __bf16* __restrict__ xb) {
    int idx = blockIdx.x * blockDim.x + threadIdx.x;
    if (idx >= HWPX * C_CH) return;
    int p = idx / C_CH, c = idx % C_CH;
    xb[idx] = (__bf16)x[(size_t)c * HWPX + p];
}

// Pack a (128x128) weight into A-fragment order:
//  idx = ((mt*4+kt)*32 + lane)*16 + i ;
//  element = w[mt*16 + lane%16][kt*32 + 16*(lane/16) + i]
__global__ void k_pack_w(const float* __restrict__ w, __bf16* __restrict__ wp) {
    int idx = blockIdx.x * blockDim.x + threadIdx.x;
    if (idx >= 8 * 4 * 32 * 16) return;
    int i  = idx & 15;
    int l  = (idx >> 4) & 31;
    int kt = (idx >> 9) & 3;
    int mt = idx >> 11;
    int o = mt * 16 + (l & 15);
    int c = kt * 32 + 16 * (l >> 4) + i;
    wp[idx] = (__bf16)w[(size_t)o * C_CH + c];
}

// Pack 3x3 conv weights (O,I,3,3) into per-tap A-fragment order.
__global__ void k_pack_wc(const float* __restrict__ wc, __bf16* __restrict__ wp) {
    int idx = blockIdx.x * blockDim.x + threadIdx.x;
    if (idx >= 9 * 8 * 4 * 32 * 16) return;
    int i   = idx & 15;
    int l   = (idx >> 4) & 31;
    int kt  = (idx >> 9) & 3;
    int mt  = (idx >> 11) & 7;
    int tap = idx >> 14;
    int dy = tap / 3, dx = tap % 3;
    int o = mt * 16 + (l & 15);
    int c = kt * 32 + 16 * (l >> 4) + i;
    wp[idx] = (__bf16)wc[(((size_t)o * C_CH + c) * 3 + dy) * 3 + dx];
}

// ---------------------------------------------------------------------------
// 1x1 conv GEMM:  e = PReLU(W(128x128) @ X(128x9216) + b)
// block = 256 (8 waves); wave mt owns out-chan tile; block owns 16 pixels.
// MODE 0: out (HW,C) bf16, packed b128 store (for Q/K)
// MODE 1: out (C,HW) bf16, b16 scatter (for V)
// ---------------------------------------------------------------------------
template <int MODE>
__global__ void __launch_bounds__(256) k_gemm1x1(const __bf16* __restrict__ xb,
                                                 const __bf16* __restrict__ wp,
                                                 const float* __restrict__ bias,
                                                 const float* __restrict__ aP,
                                                 __bf16* __restrict__ outp) {
    int lane = threadIdx.x & 31;
    int mt   = threadIdx.x >> 5;
    int p0   = blockIdx.x * 16;
    int half = lane >> 4, col = lane & 15;

    v8f acc = {0.f,0.f,0.f,0.f,0.f,0.f,0.f,0.f};
    #pragma unroll
    for (int kt = 0; kt < 4; ++kt) {
        v16bf a = ld_frag_contig(wp + (((mt << 2) + kt) << 9) + lane * 16);
        v16bf b = ld_frag_contig(xb + (size_t)(p0 + col) * C_CH + kt * 32 + (half << 4));
        acc = wmma_bf16(a, b, acc);
    }
    float alpha = aP[0];
    int p  = p0 + col;
    int m0 = mt * 16 + 8 * half;          // rows m0..m0+7 contiguous over r
    if (MODE == 0) {
        Pack8U st;
        #pragma unroll
        for (int r = 0; r < 8; ++r) {
            float v = acc[r] + bias[m0 + r];
            st.h[r] = (__bf16)((v >= 0.f) ? v : alpha * v);
        }
        *(uint4*)(outp + (size_t)p * C_CH + m0) = st.q;     // one b128 / lane
    } else {
        #pragma unroll
        for (int r = 0; r < 8; ++r) {
            float v = acc[r] + bias[m0 + r];
            v = (v >= 0.f) ? v : alpha * v;
            outp[(size_t)(m0 + r) * HWPX + p] = (__bf16)v;  // (C,HW)
        }
    }
}

// ---------------------------------------------------------------------------
// Flash attention: 1 wave per 16-query tile, streams all 9216 keys in
// 32-key chunks.  score WMMAs -> online softmax -> P staged via 1KB LDS
// into A-fragment layout -> P@V WMMAs into 8 f32 accumulators (C=128).
// Row sums come from a 9th WMMA against an all-ones B tile (P @ 1-vector),
// so no cross-lane sum shuffles are needed.  Single wave per SIMD target
// (launch_bounds 32,1): ~250 live VGPRs, no spills.  All 8 K fragments are
// loaded in one batch before the 8 QK WMMAs so the scheduler can use
// partial s_wait_loadcnt values (same pattern it found for the P@V phase).
// ---------------------------------------------------------------------------
__global__ void __launch_bounds__(32, 1)
k_flash(const __bf16* __restrict__ qb,
        const __bf16* __restrict__ kbuf,
        const __bf16* __restrict__ vbuf,
        __bf16* __restrict__ attp) {
    __shared__ __attribute__((aligned(16))) __bf16 lp[16 * 32];

    int lane = threadIdx.x & 31;
    int half = lane >> 4, col = lane & 15;
    int q0 = blockIdx.x * 16;

    // Q A-fragments (row = query pixel, K = channel), kept resident.
    v16bf qf[4];
    #pragma unroll
    for (int kt = 0; kt < 4; ++kt) {
        const __bf16* p = qb + (size_t)(q0 + col) * C_CH + kt * 32 + 16 * half;
        qf[kt] = ld_frag_contig(p);
    }

    // all-ones B tile: P @ ones == per-row sum of P in every output column
    v16bf ones;
    #pragma unroll
    for (int i = 0; i < 16; ++i) ones[i] = (__bf16)1.0f;

    v8f accO[8];
    v8f accS = {0.f,0.f,0.f,0.f,0.f,0.f,0.f,0.f};   // running row sums
    float mrow[8];
    #pragma unroll
    for (int t = 0; t < 8; ++t) {
        accO[t] = (v8f){0.f,0.f,0.f,0.f,0.f,0.f,0.f,0.f};
        mrow[t] = -3.0e38f;
    }

    for (int kb0 = 0; kb0 < HWPX; kb0 += 32) {
        // --- batch-load all 8 K B-fragments (one clause), then 8 WMMAs ---
        v16bf kf[8];
        #pragma unroll
        for (int kt = 0; kt < 4; ++kt) {
            kf[kt]     = ld_frag_contig(kbuf + (size_t)(kb0 + col) * C_CH + kt * 32 + (half << 4));
            kf[4 + kt] = ld_frag_contig(kbuf + (size_t)(kb0 + 16 + col) * C_CH + kt * 32 + (half << 4));
        }
        v8f s0 = {0.f,0.f,0.f,0.f,0.f,0.f,0.f,0.f};
        v8f s1 = {0.f,0.f,0.f,0.f,0.f,0.f,0.f,0.f};
        #pragma unroll
        for (int kt = 0; kt < 4; ++kt) {
            s0 = wmma_bf16(qf[kt], kf[kt], s0);
            s1 = wmma_bf16(qf[kt], kf[4 + kt], s1);
        }
        if (kb0 + 32 < HWPX) {  // hint next chunk toward the caches
            __builtin_prefetch(kbuf + (size_t)(kb0 + 32 + col) * C_CH, 0, 0);
            __builtin_prefetch(vbuf + (size_t)col * HWPX + kb0 + 32, 0, 0);
        }

        // --- online softmax update (rows live across 16-lane halves) ---
        #pragma unroll
        for (int r = 0; r < 8; ++r) {
            float t  = redmax16(fmaxf(s0[r], s1[r]));
            float mn = fmaxf(mrow[r], t);
            float sc = __expf(mrow[r] - mn);
            float p0v = __expf(s0[r] - mn);
            float p1v = __expf(s1[r] - mn);
            mrow[r] = mn;
            accS[r] *= sc;
            #pragma unroll
            for (int ct = 0; ct < 8; ++ct) accO[ct][r] *= sc;
            int m = r + 8 * half;                 // C/D row for this lane-half
            lp[m * 32 + col]      = (__bf16)p0v;  // P[m][key 0..15]
            lp[m * 32 + 16 + col] = (__bf16)p1v;  // P[m][key 16..31]
        }
        asm volatile("s_wait_dscnt 0" ::: "memory");  // LDS stores visible (single wave)

        // --- P as A-fragment, then P @ V (8 chan tiles) + P @ 1 (row sums) ---
        const __bf16* pp = lp + col * 32 + 16 * half;
        v16bf pa = ld_frag_contig(pp);
        #pragma unroll
        for (int ct = 0; ct < 8; ++ct) {
            v16bf vf = ld_frag_contig(vbuf + (size_t)(ct * 16 + col) * HWPX + kb0 + (half << 4));
            accO[ct] = wmma_bf16(pa, vf, accO[ct]);
        }
        accS = wmma_bf16(pa, ones, accS);
    }

    // --- epilogue: 1/l scaling, write into padded NHWC bf16 buffer ---
    // A 16-query tile lies inside one image row (96 % 16 == 0).
    int h = q0 / WIMG, wb = q0 % WIMG;
    size_t base = ((size_t)(h + 1) * PADW + (wb + 1)) * C_CH;
    float inv[8];
    #pragma unroll
    for (int r = 0; r < 8; ++r) inv[r] = 1.0f / accS[r];
    #pragma unroll
    for (int ct = 0; ct < 8; ++ct) {
        #pragma unroll
        for (int r = 0; r < 8; ++r) {
            attp[base + (size_t)(r + 8 * half) * C_CH + ct * 16 + col] =
                (__bf16)(accO[ct][r] * inv[r]);
        }
    }
}

// ---------------------------------------------------------------------------
// 3x3 conv (implicit GEMM over padded NHWC) + bias + exact GELU.
// LAST 0 : write next x as (HW,C) bf16 (packed b128 store)
// LAST 1 : out = x_orig + res, NCHW f32 with fused residual
// ---------------------------------------------------------------------------
template <int LAST>
__global__ void __launch_bounds__(256) k_conv3(const __bf16* __restrict__ attp,
                                               const __bf16* __restrict__ wcp,
                                               const float* __restrict__ bc,
                                               const float* __restrict__ xorig,
                                               __bf16* __restrict__ xnext,
                                               float* __restrict__ outp) {
    int lane = threadIdx.x & 31;
    int mt   = threadIdx.x >> 5;
    int half = lane >> 4, col = lane & 15;
    int h  = blockIdx.x / 6;
    int w0 = (blockIdx.x % 6) * 16;

    v8f acc = {0.f,0.f,0.f,0.f,0.f,0.f,0.f,0.f};
    #pragma unroll
    for (int tap = 0; tap < 9; ++tap) {
        int dy = tap / 3, dx = tap % 3;
        #pragma unroll
        for (int kt = 0; kt < 4; ++kt) {
            v16bf a = ld_frag_contig(wcp + ((((tap * 8 + mt) << 2) + kt) << 9) + lane * 16);
            v16bf b = ld_frag_contig(attp + ((size_t)(h + dy) * PADW + (w0 + col + dx)) * C_CH
                                          + kt * 32 + (half << 4));
            acc = wmma_bf16(a, b, acc);
        }
    }
    int p  = h * WIMG + w0 + col;
    int m0 = mt * 16 + 8 * half;
    if (LAST == 0) {
        Pack8U st;
        #pragma unroll
        for (int r = 0; r < 8; ++r) {
            float v = acc[r] + bc[m0 + r];
            st.h[r] = (__bf16)(0.5f * v * (1.0f + erff(v * 0.70710678118654752f)));
        }
        *(uint4*)(xnext + (size_t)p * C_CH + m0) = st.q;    // one b128 / lane
    } else {
        #pragma unroll
        for (int r = 0; r < 8; ++r) {
            float v = acc[r] + bc[m0 + r];
            float g = 0.5f * v * (1.0f + erff(v * 0.70710678118654752f));
            size_t o = (size_t)(m0 + r) * HWPX + p;         // NCHW f32
            outp[o] = xorig[o] + g;
        }
    }
}

// ---------------------------------------------------------------------------

static inline size_t alignup(size_t x) { return (x + 255) & ~(size_t)255; }

extern "C" void kernel_launch(void* const* d_in, const int* in_sizes, int n_in,
                              void* d_out, int out_size, void* d_ws, size_t ws_size,
                              hipStream_t stream) {
    (void)in_sizes; (void)n_in; (void)out_size; (void)ws_size;
    const float* x  = (const float*)d_in[0];
    const float* w1 = (const float*)d_in[1];
    const float* b1 = (const float*)d_in[2];
    const float* a1 = (const float*)d_in[3];
    const float* w2 = (const float*)d_in[4];
    const float* b2 = (const float*)d_in[5];
    const float* a2 = (const float*)d_in[6];
    const float* w3 = (const float*)d_in[7];
    const float* b3 = (const float*)d_in[8];
    const float* a3 = (const float*)d_in[9];
    const float* wc = (const float*)d_in[10];
    const float* bc = (const float*)d_in[11];
    float* out = (float*)d_out;

    // workspace carve (~14.6 MB)
    char* ws = (char*)d_ws;
    size_t off = 0;
    const size_t actB = (size_t)HWPX * C_CH * sizeof(__bf16);   // 2.36 MB
    __bf16* xbA  = (__bf16*)(ws + off); off = alignup(off + actB);
    __bf16* xbB  = (__bf16*)(ws + off); off = alignup(off + actB);
    __bf16* qbuf = (__bf16*)(ws + off); off = alignup(off + actB);
    __bf16* kbuf = (__bf16*)(ws + off); off = alignup(off + actB);
    __bf16* vbuf = (__bf16*)(ws + off); off = alignup(off + actB);
    const size_t padB = (size_t)PADW * PADW * C_CH * sizeof(__bf16);
    __bf16* attp = (__bf16*)(ws + off); off = alignup(off + padB);
    const size_t wpB = (size_t)8 * 4 * 32 * 16 * sizeof(__bf16);
    __bf16* wp1 = (__bf16*)(ws + off); off = alignup(off + wpB);
    __bf16* wp2 = (__bf16*)(ws + off); off = alignup(off + wpB);
    __bf16* wp3 = (__bf16*)(ws + off); off = alignup(off + wpB);
    __bf16* wcp = (__bf16*)(ws + off); off = alignup(off + 9 * wpB);

    // prep: zero padded att buffer (borders stay zero), convert x, pack weights
    hipMemsetAsync(attp, 0, padB, stream);
    k_convert_x<<<(HWPX * C_CH + 255) / 256, 256, 0, stream>>>(x, xbA);
    k_pack_w<<<64, 256, 0, stream>>>(w1, wp1);
    k_pack_w<<<64, 256, 0, stream>>>(w2, wp2);
    k_pack_w<<<64, 256, 0, stream>>>(w3, wp3);
    k_pack_wc<<<576, 256, 0, stream>>>(wc, wcp);

    // 3 depth units, shared weights, ping-pong activations
    for (int d = 0; d < 3; ++d) {
        const __bf16* xin = (d & 1) ? xbB : xbA;
        __bf16* xnx       = (d & 1) ? xbA : xbB;
        k_gemm1x1<0><<<576, 256, 0, stream>>>(xin, wp1, b1, a1, qbuf);  // Q (HW,C)
        k_gemm1x1<0><<<576, 256, 0, stream>>>(xin, wp2, b2, a2, kbuf);  // K (HW,C)
        k_gemm1x1<1><<<576, 256, 0, stream>>>(xin, wp3, b3, a3, vbuf);  // V (C,HW)
        k_flash<<<576, 32, 0, stream>>>(qbuf, kbuf, vbuf, attp);
        if (d == 2) k_conv3<1><<<576, 256, 0, stream>>>(attp, wcp, bc, x, nullptr, out);
        else        k_conv3<0><<<576, 256, 0, stream>>>(attp, wcp, bc, x, xnx, nullptr);
    }
}
